// Conv2D_62431644615223
// MI455X (gfx1250) — matrix operand order repaired and checked by reading, not compiled, not run on hardware
//
#include <hip/hip_runtime.h>

typedef __attribute__((ext_vector_type(16))) __bf16        v16bf;
typedef __attribute__((ext_vector_type(8)))  float         v8f;
typedef __attribute__((ext_vector_type(4)))  unsigned int  u32x4;
typedef __attribute__((ext_vector_type(4)))  float         f32x4;

#define NB    32
#define H_    128
#define W_    128
#define CIN   128
#define COUT  128
#define KH    3
#define KW    3
#define HO    126
#define WO    126
#define MTOT  (HO*WO)          /* 15876 */
#define MT    64               /* M tile per block */
#define KC    32               /* reduction chunk = WMMA K */
#define KRED  (KH*KW*CIN)      /* 1152 */

union Frag { v16bf v; u32x4 q[2]; };

// Split fp32 into bf16 hi (truncation -> exact residual) + bf16 lo (RNE).
__device__ __forceinline__ void split_bf16(float x, unsigned short& hi, unsigned short& lo) {
    unsigned u = __builtin_bit_cast(unsigned, x);
    hi = (unsigned short)(u >> 16);
    float hf = __builtin_bit_cast(float, u & 0xFFFF0000u);
    float r  = x - hf;
    unsigned v = __builtin_bit_cast(unsigned, r);
    unsigned rnd = v + 0x7FFFu + ((v >> 16) & 1u);
    lo = (unsigned short)(rnd >> 16);
}

__global__ __launch_bounds__(256)
void conv2d_persample_wmma_bf16x2(const float* __restrict__ X,
                                  const float* __restrict__ Kr,
                                  float* __restrict__ Out)
{
    __shared__ __align__(16) unsigned short Ah[MT * KC];     // 4 KB
    __shared__ __align__(16) unsigned short Al[MT * KC];     // 4 KB
    __shared__ __align__(16) unsigned short Bh[COUT * KC];   // 8 KB (transposed: [co][ci])
    __shared__ __align__(16) unsigned short Bl[COUT * KC];   // 8 KB

    const int tileM  = blockIdx.x;
    const int b      = blockIdx.y;
    const int t      = threadIdx.x;
    const int lane   = t & 31;
    const int wid    = t >> 5;
    const int waveM  = wid & 3;     // 0..3 -> 16-row strip
    const int waveN  = wid >> 2;    // 0..1 -> 64-col strip
    const int half   = (lane < 16) ? 0 : 1;
    const int lrow   = lane & 15;

    const float* Xb = X  + (size_t)b * H_ * W_ * CIN;
    const float* Kb = Kr + (size_t)b * KH * KW * CIN * COUT;

    v8f acc[4] = {v8f{}, v8f{}, v8f{}, v8f{}};

    for (int kc = 0; kc < KRED; kc += KC) {
        const int kh  = kc / (KW * CIN);
        const int rem = kc - kh * (KW * CIN);
        const int kw  = rem / CIN;
        const int ci0 = rem - kw * CIN;

        // ---- stage A tile (MT x KC) as bf16 hi/lo (im2col gather) ----
        #pragma unroll
        for (int i = 0; i < 2; ++i) {
            int q  = t + i * 256;            // 0..511 float4 slots
            int m  = q >> 3;
            int c4 = (q & 7) << 2;
            int gp = tileM * MT + m;
            if (gp > MTOT - 1) gp = MTOT - 1;           // clamp: loads stay valid
            int h = gp / WO, w = gp - h * WO;
            const f32x4 x = *(const f32x4*)&Xb[(((h + kh) * W_) + (w + kw)) * CIN + ci0 + c4];
            #pragma unroll
            for (int j = 0; j < 4; ++j) {
                unsigned short hi, lo;
                split_bf16(x[j], hi, lo);
                Ah[m * KC + c4 + j] = hi;
                Al[m * KC + c4 + j] = lo;
            }
        }

        // ---- stage B tile (KC x COUT) -> transposed LDS [co][ci] ----
        #pragma unroll
        for (int i = 0; i < 4; ++i) {
            int q  = t + i * 256;            // 0..1023 float4 slots
            int r  = q >> 5;                 // ci within chunk
            int c4 = (q & 31) << 2;          // co
            const float* src = &Kb[(((kh * KW + kw) * CIN) + ci0 + r) * COUT + c4];
            const f32x4 x = *(const f32x4*)src;
            #pragma unroll
            for (int j = 0; j < 4; ++j) {
                unsigned short hi, lo;
                split_bf16(x[j], hi, lo);
                Bh[(c4 + j) * KC + r] = hi;
                Bl[(c4 + j) * KC + r] = lo;
            }
            if (kc + KC < KRED)
                __builtin_prefetch(src + (size_t)KC * COUT, 0, 1);  // global_prefetch_b8
        }
        __syncthreads();

        // ---- WMMA: A fragment (16x32 bf16) per doc layout ----
        // lane<16: M=lrow, K{0..7,16..23}; lane>=16: K{8..15,24..31}
        Frag ah, al;
        {
            const unsigned short* pa = &Ah[(waveM * 16 + lrow) * KC + half * 8];
            ah.q[0] = *(const u32x4*)pa;
            ah.q[1] = *(const u32x4*)(pa + 16);
            const unsigned short* pb = &Al[(waveM * 16 + lrow) * KC + half * 8];
            al.q[0] = *(const u32x4*)pb;
            al.q[1] = *(const u32x4*)(pb + 16);
        }

        // B fragment (32x16 bf16): lane<16 holds K{0..7,16..23} of col=lrow, lane>=16 K{8..15,24..31}
        #pragma unroll
        for (int nt = 0; nt < 4; ++nt) {
            const int col = waveN * 64 + nt * 16 + lrow;
            Frag bh, bl;
            const unsigned short* pb = &Bh[col * KC + half * 8];
            bh.q[0] = *(const u32x4*)pb;
            bh.q[1] = *(const u32x4*)(pb + 16);
            const unsigned short* pl = &Bl[col * KC + half * 8];
            bl.q[0] = *(const u32x4*)pl;
            bl.q[1] = *(const u32x4*)(pl + 16);

            acc[nt] = __builtin_amdgcn_wmma_f32_16x16x32_bf16(
                false, ah.v, false, bh.v, (short)0, acc[nt], false, false);
            acc[nt] = __builtin_amdgcn_wmma_f32_16x16x32_bf16(
                false, ah.v, false, bl.v, (short)0, acc[nt], false, false);
            acc[nt] = __builtin_amdgcn_wmma_f32_16x16x32_bf16(
                false, al.v, false, bh.v, (short)0, acc[nt], false, false);
        }
        __syncthreads();
    }

    // ---- epilogue: C layout VGPR r -> M=r (lanes 0-15) / M=r+8 (lanes 16-31), N=lane%16
    const int mBase = tileM * MT + waveM * 16 + half * 8;
    #pragma unroll
    for (int nt = 0; nt < 4; ++nt) {
        const int co = waveN * 64 + nt * 16 + lrow;
        #pragma unroll
        for (int r = 0; r < 8; ++r) {
            int gp = mBase + r;
            if (gp < MTOT) {
                int h = gp / WO, w = gp - h * WO;
                Out[(((size_t)b * HO + h) * WO + w) * COUT + co] = acc[nt][r];
            }
        }
    }
}

extern "C" void kernel_launch(void* const* d_in, const int* in_sizes, int n_in,
                              void* d_out, int out_size, void* d_ws, size_t ws_size,
                              hipStream_t stream) {
    const float* X  = (const float*)d_in[0];
    const float* Kr = (const float*)d_in[1];
    float*       O  = (float*)d_out;
    dim3 grid((MTOT + MT - 1) / MT, NB);   // 249 x 32 blocks
    conv2d_persample_wmma_bf16x2<<<grid, dim3(256), 0, stream>>>(X, Kr, O);
}
